// GCN2Layers_4329327034972
// MI455X (gfx1250) — matrix operand order repaired and checked by reading, not compiled, not run on hardware
//
#include <hip/hip_runtime.h>

typedef __attribute__((ext_vector_type(16))) __bf16 v16bf;
typedef __attribute__((ext_vector_type(8)))  __bf16 v8bf;
typedef __attribute__((ext_vector_type(8)))  float  v8f;

#define N_NODES 100000
#define N_EDGES 1600000
#define D_IN    256
#define D_HID   128
#define D_OUT   64

// float -> bf16 round-to-nearest-even (used in pack/convert and combine1 kernels)
__device__ __forceinline__ __bf16 f2bf(float f) {
    unsigned u = __builtin_bit_cast(unsigned, f);
    u += 0x7FFFu + ((u >> 16) & 1u);
    unsigned short h = (unsigned short)(u >> 16);
    return __builtin_bit_cast(__bf16, h);
}

__device__ __forceinline__ v16bf join16(v8bf lo, v8bf hi) {
    union { v16bf v; v8bf h[2]; } u;
    u.h[0] = lo; u.h[1] = hi;
    return u.v;
}

// ---------------------------------------------------------------------------
// Utility / prep kernels
// ---------------------------------------------------------------------------
__global__ void fill_f32(float* __restrict__ p, float v, int n) {
    int i = blockIdx.x * blockDim.x + threadIdx.x;
    if (i < n) p[i] = v;
}

__global__ void deg_kernel(const long long* __restrict__ dst, float* __restrict__ deg) {
    int e = blockIdx.x * blockDim.x + threadIdx.x;
    if (e < N_EDGES) unsafeAtomicAdd(&deg[(int)dst[e]], 1.0f);
}

__global__ void rsqrt_kernel(const float* __restrict__ deg, float* __restrict__ dinv) {
    int i = blockIdx.x * blockDim.x + threadIdx.x;
    if (i < N_NODES) dinv[i] = rsqrtf(deg[i]);
}

// row-major f32 -> row-major bf16
__global__ void cvt_bf16_kernel(const float* __restrict__ src, __bf16* __restrict__ dst, int n) {
    int i = blockIdx.x * blockDim.x + threadIdx.x;
    if (i < n) dst[i] = f2bf(src[i]);
}

// Pack W[K,N] (row-major f32) into WMMA B-fragment order (bf16):
//   Wp[((ntile*KT + ktile)*32 + lane)*16 + h]  where lane = halfg*16 + l15,
//   h < 8 -> W[ktile*32 + halfg*8 + h][ntile*16 + l15], h >= 8 -> W[ktile*32 + 16 + halfg*8 + h-8][ntile*16 + l15]
__global__ void pack_w_kernel(const float* __restrict__ W, __bf16* __restrict__ Wp,
                              int K, int N) {
    int i = blockIdx.x * blockDim.x + threadIdx.x;
    if (i >= K * N) return;
    int h     = i & 15;
    int lane  = (i >> 4) & 31;
    int blk   = i >> 9;                 // ntile*KT + ktile
    int KT    = K >> 5;
    int ntile = blk / KT;
    int ktile = blk % KT;
    int halfg = lane >> 4;
    int l15   = lane & 15;
    int k = ktile * 32 + (h < 8 ? halfg * 8 + h : 16 + halfg * 8 + (h - 8));
    int n = ntile * 16 + l15;
    Wp[i] = f2bf(W[(size_t)k * N + n]);
}

// ---------------------------------------------------------------------------
// GEMM1: H[100000,128] = Xbf[100000,256] @ W1p   (bf16 WMMA, f32 accum)
// Block = 256 threads = 8 waves; block owns one 16-row M tile, wave w owns
// N tile w*16. A fragment = 2x b128 contiguous; B fragment = 2x b128 packed.
// ---------------------------------------------------------------------------
__global__ __launch_bounds__(256)
void gemm1_kernel(const __bf16* __restrict__ Xbf, const __bf16* __restrict__ Wp,
                  float* __restrict__ H) {
    const int lane = threadIdx.x & 31;
    const int wave = threadIdx.x >> 5;
    const int m0   = blockIdx.x * 16;
    const int half = lane >> 4;
    const int l15  = lane & 15;

    const __bf16* Arow = Xbf + (size_t)(m0 + l15) * D_IN + half * 8;
    const __bf16* Bp   = Wp + ((size_t)wave * (D_IN / 32) * 32 + lane) * 16;
    const int     bcol = wave * 16 + l15;

    v8f c = {};
#pragma unroll
    for (int kt = 0; kt < D_IN / 32; ++kt) {
        v8bf alo = *(const v8bf*)(Arow + kt * 32);
        v8bf ahi = *(const v8bf*)(Arow + kt * 32 + 16);
        v8bf blo = *(const v8bf*)(Bp + (size_t)kt * 32 * 16);
        v8bf bhi = *(const v8bf*)(Bp + (size_t)kt * 32 * 16 + 8);
        c = __builtin_amdgcn_wmma_f32_16x16x32_bf16(false, join16(alo, ahi),
                                                    false, join16(blo, bhi),
                                                    (short)0, c, false, false);
    }
#pragma unroll
    for (int v = 0; v < 8; ++v)
        H[(size_t)(m0 + v + half * 8) * D_HID + bcol] = c[v];
}

// ---------------------------------------------------------------------------
// GEMM2: H2[100000,64] = H1bf[100000,128] @ W2p
// Block = 256 threads = 8 waves; waves 0-3 -> M tile 0 (N tiles 0..3),
// waves 4-7 -> M tile 1.
// ---------------------------------------------------------------------------
__global__ __launch_bounds__(256)
void gemm2_kernel(const __bf16* __restrict__ H1bf, const __bf16* __restrict__ Wp,
                  float* __restrict__ H2) {
    const int lane = threadIdx.x & 31;
    const int wave = threadIdx.x >> 5;
    const int m0   = blockIdx.x * 32 + (wave >> 2) * 16;
    const int nt   = wave & 3;
    const int half = lane >> 4;
    const int l15  = lane & 15;

    const __bf16* Arow = H1bf + (size_t)(m0 + l15) * D_HID + half * 8;
    const __bf16* Bp   = Wp + ((size_t)nt * (D_HID / 32) * 32 + lane) * 16;
    const int     bcol = nt * 16 + l15;

    v8f c = {};
#pragma unroll
    for (int kt = 0; kt < D_HID / 32; ++kt) {
        v8bf alo = *(const v8bf*)(Arow + kt * 32);
        v8bf ahi = *(const v8bf*)(Arow + kt * 32 + 16);
        v8bf blo = *(const v8bf*)(Bp + (size_t)kt * 32 * 16);
        v8bf bhi = *(const v8bf*)(Bp + (size_t)kt * 32 * 16 + 8);
        c = __builtin_amdgcn_wmma_f32_16x16x32_bf16(false, join16(alo, ahi),
                                                    false, join16(blo, bhi),
                                                    (short)0, c, false, false);
    }
#pragma unroll
    for (int v = 0; v < 8; ++v)
        H2[(size_t)(m0 + v + half * 8) * D_OUT + bcol] = c[v];
}

// ---------------------------------------------------------------------------
// Edge scatter: agg[dst] += h[src] * dinv[src]*dinv[dst]
// One wave per edge; lane-strided features -> coalesced loads + f32 atomics
// (h and agg both fit in the 192MB L2).
// ---------------------------------------------------------------------------
__global__ __launch_bounds__(256)
void scatter_kernel(const float* __restrict__ H,
                    const long long* __restrict__ src,
                    const long long* __restrict__ dst,
                    const float* __restrict__ dinv,
                    float* __restrict__ Agg, int D) {
    int e = blockIdx.x * 8 + (threadIdx.x >> 5);
    if (e >= N_EDGES) return;
    int lane = threadIdx.x & 31;
    int s = (int)src[e];
    int d = (int)dst[e];
    float norm = dinv[s] * dinv[d];
    const float* hs = H + (size_t)s * D;
    float*       ad = Agg + (size_t)d * D;
    for (int f = lane; f < D; f += 32)
        unsafeAtomicAdd(&ad[f], hs[f] * norm);
}

// ---------------------------------------------------------------------------
// Combine layer 1: h1 = relu(agg + h*dinv^2 + b1)  -> bf16 (GEMM2 input)
// ---------------------------------------------------------------------------
__global__ void combine1_kernel(const float* __restrict__ Agg,
                                const float* __restrict__ H,
                                const float* __restrict__ dinv,
                                const float* __restrict__ bias,
                                __bf16* __restrict__ Out) {
    int i = blockIdx.x * blockDim.x + threadIdx.x;
    if (i >= N_NODES * D_HID) return;
    int node = i >> 7;           // / 128
    int f    = i & (D_HID - 1);
    float di = dinv[node];
    float v  = Agg[i] + H[i] * di * di + bias[f];
    Out[i] = f2bf(fmaxf(v, 0.0f));
}

// Combine layer 2: out = agg + h*dinv^2 + b2  -> f32 (final output)
__global__ void combine2_kernel(const float* __restrict__ Agg,
                                const float* __restrict__ H,
                                const float* __restrict__ dinv,
                                const float* __restrict__ bias,
                                float* __restrict__ Out) {
    int i = blockIdx.x * blockDim.x + threadIdx.x;
    if (i >= N_NODES * D_OUT) return;
    int node = i >> 6;           // / 64
    int f    = i & (D_OUT - 1);
    float di = dinv[node];
    Out[i] = Agg[i] + H[i] * di * di + bias[f];
}

// ---------------------------------------------------------------------------
extern "C" void kernel_launch(void* const* d_in, const int* in_sizes, int n_in,
                              void* d_out, int out_size, void* d_ws, size_t ws_size,
                              hipStream_t stream) {
    const float*     x   = (const float*)d_in[0];
    const long long* ei  = (const long long*)d_in[1];    // int64 [2, N_EDGES]
    const float*     W1  = (const float*)d_in[2];
    const float*     b1  = (const float*)d_in[3];
    const float*     W2  = (const float*)d_in[4];
    const float*     b2  = (const float*)d_in[5];
    float*           out = (float*)d_out;

    const long long* srcI = ei;
    const long long* dstI = ei + N_EDGES;

    // Workspace partition (all 16B-aligned by construction)
    char* p = (char*)d_ws;
    float*  h_lin1 = (float*)p;  p += (size_t)N_NODES * D_HID * 4;   // 51.2 MB
    float*  agg1   = (float*)p;  p += (size_t)N_NODES * D_HID * 4;   // 51.2 MB
    float*  h_lin2 = (float*)p;  p += (size_t)N_NODES * D_OUT * 4;   // 25.6 MB
    float*  agg2   = (float*)p;  p += (size_t)N_NODES * D_OUT * 4;   // 25.6 MB
    float*  deg    = (float*)p;  p += (size_t)N_NODES * 4;
    float*  dinv   = (float*)p;  p += (size_t)N_NODES * 4;
    __bf16* xbf    = (__bf16*)p; p += (size_t)N_NODES * D_IN * 2;    // 51.2 MB
    __bf16* h1bf   = (__bf16*)p; p += (size_t)N_NODES * D_HID * 2;   // 25.6 MB
    __bf16* W1p    = (__bf16*)p; p += (size_t)D_IN * D_HID * 2;      // 64 KB
    __bf16* W2p    = (__bf16*)p; p += (size_t)D_HID * D_OUT * 2;     // 16 KB

    // 1. init + degree
    fill_f32<<<(N_NODES + 255) / 256, 256, 0, stream>>>(deg, 1.0f, N_NODES);
    fill_f32<<<(N_NODES * D_HID + 255) / 256, 256, 0, stream>>>(agg1, 0.0f, N_NODES * D_HID);
    fill_f32<<<(N_NODES * D_OUT + 255) / 256, 256, 0, stream>>>(agg2, 0.0f, N_NODES * D_OUT);
    deg_kernel<<<(N_EDGES + 255) / 256, 256, 0, stream>>>(dstI, deg);
    rsqrt_kernel<<<(N_NODES + 255) / 256, 256, 0, stream>>>(deg, dinv);

    // 2. precision prep: x -> bf16, W1/W2 -> packed bf16 fragments
    cvt_bf16_kernel<<<(N_NODES * D_IN + 255) / 256, 256, 0, stream>>>(x, xbf, N_NODES * D_IN);
    pack_w_kernel<<<(D_IN * D_HID + 255) / 256, 256, 0, stream>>>(W1, W1p, D_IN, D_HID);
    pack_w_kernel<<<(D_HID * D_OUT + 255) / 256, 256, 0, stream>>>(W2, W2p, D_HID, D_OUT);

    // 3. layer 1
    gemm1_kernel<<<N_NODES / 16, 256, 0, stream>>>(xbf, W1p, h_lin1);
    scatter_kernel<<<(N_EDGES + 7) / 8, 256, 0, stream>>>(h_lin1, srcI, dstI, dinv, agg1, D_HID);
    combine1_kernel<<<(N_NODES * D_HID + 255) / 256, 256, 0, stream>>>(
        agg1, h_lin1, dinv, b1, h1bf);

    // 4. layer 2
    gemm2_kernel<<<N_NODES / 32, 256, 0, stream>>>(h1bf, W2p, h_lin2);
    scatter_kernel<<<(N_EDGES + 7) / 8, 256, 0, stream>>>(h_lin2, srcI, dstI, dinv, agg2, D_OUT);
    combine2_kernel<<<(N_NODES * D_OUT + 255) / 256, 256, 0, stream>>>(
        agg2, h_lin2, dinv, b2, out);
}
